// FusedTimeGroupAttentionTriton_29059748724880
// MI455X (gfx1250) — matrix operand order repaired and checked by reading, not compiled, not run on hardware
//
#include <hip/hip_runtime.h>
#include <hip/hip_bf16.h>

typedef _Float16 h16;
typedef __attribute__((ext_vector_type(8)))  _Float16 v8h;
typedef __attribute__((ext_vector_type(16))) _Float16 v16h;
typedef __attribute__((ext_vector_type(8)))  float    v8f;

#define DMODEL 768
#define NHEAD  12
#define DKV    64
#define BB     16
#define SS     512
#define MROWS  (BB * SS) /* 8192 */

#if defined(__HIP_DEVICE_COMPILE__) && __has_builtin(__builtin_amdgcn_tensor_load_to_lds)
#define USE_TDM 1
#else
#define USE_TDM 0
#endif

// ---------------- WMMA fragment loaders (CDNA5 16x16x32 f16 layouts) ----------------
// A (16x32, MxK): lane m = L%16; halves 0-7 -> K = 8*(L/16)+i ; halves 8-15 -> +16
__device__ __forceinline__ v16h frag_ld_a(const h16* p, int ld) {
  int lane = threadIdx.x & 31;
  const h16* q = p + (lane & 15) * ld + ((lane >> 4) << 3);
  v8h lo = *(const v8h*)(q);
  v8h hi = *(const v8h*)(q + 16);
  v16h r;
#pragma unroll
  for (int i = 0; i < 8; ++i) { r[i] = lo[i]; r[i + 8] = hi[i]; }
  return r;
}
// B (32x16, KxN) loaded from N-major memory (rows = n): lane n = L%16;
// halves 0-7 -> K = 8*(L/16)+i ; halves 8-15 -> +16
__device__ __forceinline__ v16h frag_ld_b(const h16* p, int ld) {
  int lane = threadIdx.x & 31;
  const h16* q = p + (lane & 15) * ld + ((lane >> 4) << 3);
  v8h lo = *(const v8h*)(q);
  v8h hi = *(const v8h*)(q + 16);
  v16h r;
#pragma unroll
  for (int i = 0; i < 8; ++i) { r[i] = lo[i]; r[i + 8] = hi[i]; }
  return r;
}

__device__ __forceinline__ v8f wmma16(v16h a, v16h b, v8f c) {
  return __builtin_amdgcn_wmma_f32_16x16x32_f16(false, a, false, b, (short)0, c,
                                                false, false);
}

#if USE_TDM
// ---------------- Tensor Data Mover: 2D tile (rows x row_elems of f16) -> LDS ----------------
typedef __attribute__((ext_vector_type(4))) unsigned tdm_v4u;
typedef __attribute__((ext_vector_type(8))) int      tdm_v8i;
typedef __attribute__((ext_vector_type(4))) int      tdm_v4i;

__device__ __forceinline__ void tdm_load_2d(const h16* gsrc, void* ldst,
                                            unsigned rows, unsigned row_elems,
                                            unsigned row_stride_elems) {
  unsigned lds_off = (unsigned)(uintptr_t)ldst;          // generic LDS addr: low 32 bits = offset
  unsigned long long ga = (unsigned long long)(uintptr_t)gsrc;
  tdm_v4u g0;
  g0[0] = 1u;                                            // count=1 valid, user mode
  g0[1] = lds_off;                                       // lds_addr
  g0[2] = (unsigned)ga;                                  // global_addr[31:0]
  g0[3] = (unsigned)((ga >> 32) & 0x01FFFFFFu) | (2u << 30); // global_addr[56:32] | type=2
  tdm_v8i g1;
  g1[0] = (int)(1u << 16);                               // data_size=1 (2 bytes/elem)
  g1[1] = (int)((row_elems & 0xFFFFu) << 16);            // tensor_dim0[15:0]
  g1[2] = (int)((row_elems >> 16) | ((rows & 0xFFFFu) << 16)); // dim0 hi | tensor_dim1 lo
  g1[3] = (int)((rows >> 16) | ((row_elems & 0xFFFFu) << 16)); // dim1 hi | tile_dim0
  g1[4] = (int)rows;                                     // tile_dim1 (tile_dim2 = 0)
  g1[5] = (int)row_stride_elems;                         // tensor_dim0_stride[31:0]
  g1[6] = 0;
  g1[7] = 0;                                             // tensor_dim1_stride unused (2D)
  tdm_v4i z4 = {0, 0, 0, 0};                             // D# groups 2/3: unused (2D tensor)
  tdm_v8i z8 = {0, 0, 0, 0, 0, 0, 0, 0};
  __builtin_amdgcn_tensor_load_to_lds(g0, g1, z4, z4, z8, 0);
}
#endif

// ---------------- elementwise: f32 -> f16 (weights) ----------------
__global__ __launch_bounds__(256) void cvt_f16(const float* __restrict__ in,
                                               h16* __restrict__ out, int n) {
  int i = blockIdx.x * 256 + threadIdx.x;
  if (i < n) out[i] = (h16)in[i];
}

// ---------------- RMSNorm row -> f16 (optional (b,s)->(s,b) row remap) ----------------
__global__ __launch_bounds__(256) void rmsnorm_f16(const float* __restrict__ x,
                                                   const float* __restrict__ w,
                                                   h16* __restrict__ out,
                                                   int transpose) {
  __shared__ float red[256];
  int r = blockIdx.x;
  int src = transpose ? ((r % BB) * SS + (r / BB)) : r;
  const float* row = x + (size_t)src * DMODEL;
  float ss = 0.f;
  for (int i = threadIdx.x; i < DMODEL; i += 256) { float v = row[i]; ss += v * v; }
  red[threadIdx.x] = ss;
  __syncthreads();
  for (int off = 128; off > 0; off >>= 1) {
    if (threadIdx.x < off) red[threadIdx.x] += red[threadIdx.x + off];
    __syncthreads();
  }
  float scale = rsqrtf(red[0] / (float)DMODEL + 1e-6f);
  for (int i = threadIdx.x; i < DMODEL; i += 256)
    out[(size_t)r * DMODEL + i] = (h16)(w[i] * row[i] * scale);
}

// ---------------- generic 8192x768x768 WMMA GEMM: C = X * W^T ----------------
__global__ __launch_bounds__(256) void gemm768(const h16* __restrict__ X,
                                               const h16* __restrict__ W,
                                               int mode, int seqlen,
                                               const float* __restrict__ resid,
                                               float* __restrict__ outf,
                                               h16* __restrict__ outh,
                                               int trans_out) {
  int w = threadIdx.x >> 5;
  int lane = threadIdx.x & 31;
  int m0 = blockIdx.x * 64 + (w & 3) * 16;
  int n0 = blockIdx.y * 128 + (w >> 2) * 64;
  v8f acc0 = {}, acc1 = {}, acc2 = {}, acc3 = {};
  const h16* xr = X + (size_t)m0 * DMODEL;
  for (int k = 0; k < DMODEL; k += 32) {
    __builtin_prefetch(xr + k + 256, 0, 3);
    v16h a = frag_ld_a(xr + k, DMODEL);
    acc0 = wmma16(a, frag_ld_b(W + (size_t)(n0 +  0) * DMODEL + k, DMODEL), acc0);
    acc1 = wmma16(a, frag_ld_b(W + (size_t)(n0 + 16) * DMODEL + k, DMODEL), acc1);
    acc2 = wmma16(a, frag_ld_b(W + (size_t)(n0 + 32) * DMODEL + k, DMODEL), acc2);
    acc3 = wmma16(a, frag_ld_b(W + (size_t)(n0 + 48) * DMODEL + k, DMODEL), acc3);
  }
  int colb = lane & 15;
  int rowoff = (lane >> 4) << 3;
#pragma unroll
  for (int j = 0; j < 4; ++j) {
    v8f acc = (j == 0) ? acc0 : (j == 1) ? acc1 : (j == 2) ? acc2 : acc3;
#pragma unroll
    for (int e = 0; e < 8; ++e) {
      int m = m0 + e + rowoff;
      int n = n0 + j * 16 + colb;
      float v = acc[e];
      if (mode == 0) {
        int orow = trans_out ? ((m % BB) * SS + (m / BB)) : m;
        outf[(size_t)orow * DMODEL + n] = resid[(size_t)orow * DMODEL + n] + v;
      } else {
        int batch = m / seqlen, s = m % seqlen, hh = n >> 6, d = n & 63;
        if (mode == 1)
          outh[(((size_t)batch * NHEAD + hh) * seqlen + s) * DKV + d] = (h16)v;
        else
          outh[(((size_t)batch * NHEAD + hh) * DKV + d) * seqlen + s] = (h16)v;
      }
    }
  }
}

// ---------------- RoPE on Q,K (in place, f16) ----------------
__global__ __launch_bounds__(256) void rope_qk(h16* __restrict__ Q, h16* __restrict__ K,
                                               const int* __restrict__ pos,
                                               const float* __restrict__ inv_freq) {
  int idx = blockIdx.x * 256 + threadIdx.x;
  int d = idx & 31;
  int s = (idx >> 5) & (SS - 1);
  int bh = idx >> 14;
  int b = bh / NHEAD;
  float p = (float)pos[b * SS + s];
  float f = p * inv_freq[d];
  float c = cosf(f), sn = sinf(f);
  size_t base = ((size_t)bh * SS + s) * DKV + d;
  float q1 = (float)Q[base], q2 = (float)Q[base + 32];
  Q[base]      = (h16)(q1 * c - q2 * sn);
  Q[base + 32] = (h16)(q2 * c + q1 * sn);
  float k1 = (float)K[base], k2 = (float)K[base + 32];
  K[base]      = (h16)(k1 * c - k2 * sn);
  K[base + 32] = (h16)(k2 * c + k1 * sn);
}

// ---------------- stage-1 attention: flash over 512 keys; K/V tiles via TDM double-buffer ----------------
__global__ __launch_bounds__(128) void attn1(const h16* __restrict__ Q,
                                             const h16* __restrict__ K,
                                             const h16* __restrict__ Vt,
                                             const float* __restrict__ mask,
                                             h16* __restrict__ O) {
  __shared__ __align__(16) float    s_sc[4][16][32];
  __shared__ __align__(16) _Float16 s_p[4][16][32];
  __shared__ __align__(16) float    s_aux[4][16];
#if USE_TDM
  __shared__ __align__(16) h16 kt[2][32][64];   // [buf][key][d]
  __shared__ __align__(16) h16 vt[2][64][32];   // [buf][d][key]
#endif
  int w = threadIdx.x >> 5, lane = threadIdx.x & 31;
  int bh = blockIdx.x;
  int q0 = blockIdx.y * 64 + w * 16;
  const h16*  Qb = Q + (size_t)bh * SS * DKV;
  const h16*  Kb = K + (size_t)bh * SS * DKV;
  const h16*  Vb = Vt + (size_t)bh * DKV * SS;  // [d][s]
  const float* Mb = mask + (size_t)bh * SS * SS;
#if USE_TDM
  if (w == 0) {  // one wave drives the Tensor Data Mover (EXEC ignored by TDM)
    tdm_load_2d(Kb,            &kt[0][0][0], 32, 64, DKV);
    tdm_load_2d(Vb,            &vt[0][0][0], 64, 32, SS);
    tdm_load_2d(Kb + 32 * DKV, &kt[1][0][0], 32, 64, DKV);
    tdm_load_2d(Vb + 32,       &vt[1][0][0], 64, 32, SS);
  }
#endif
  v16h qa0 = frag_ld_a(Qb + (size_t)q0 * DKV, DKV);
  v16h qa1 = frag_ld_a(Qb + (size_t)q0 * DKV + 32, DKV);
  v8f o0 = {}, o1 = {}, o2 = {}, o3 = {};
  float m_run = -3.0e38f, l_run = 0.f;
  int colb = lane & 15, rowoff = (lane >> 4) << 3;
  for (int kb = 0; kb < SS; kb += 32) {
    const h16* kp; const h16* vp; int kld, vld, vc0;
#if USE_TDM
    int ib = (kb >> 5) & 1;
    if (w == 0) {
      if (kb + 32 < SS) __builtin_amdgcn_s_wait_tensorcnt(2);  // pair(kb) done (in-order)
      else              __builtin_amdgcn_s_wait_tensorcnt(0);
    }
    __syncthreads();   // tile buffer ib visible to all waves
    kp = &kt[ib][0][0]; kld = 64;
    vp = &vt[ib][0][0]; vld = 32; vc0 = 0;
#else
    kp = Kb + (size_t)kb * DKV; kld = DKV;
    vp = Vb; vld = SS; vc0 = kb;
#endif
    v8f sc0 = {}, sc1 = {};
    sc0 = wmma16(qa0, frag_ld_b(kp, kld), sc0);
    sc0 = wmma16(qa1, frag_ld_b(kp + 32, kld), sc0);
    sc1 = wmma16(qa0, frag_ld_b(kp + 16 * kld, kld), sc1);
    sc1 = wmma16(qa1, frag_ld_b(kp + 16 * kld + 32, kld), sc1);
#pragma unroll
    for (int e = 0; e < 8; ++e) {
      int r = e + rowoff;
      size_t mrow = (size_t)(q0 + r) * SS + kb + colb;
      s_sc[w][r][colb]      = sc0[e] + Mb[mrow];
      s_sc[w][r][colb + 16] = sc1[e] + Mb[mrow + 16];
    }
    __syncthreads();
    if (lane < 16) {
      float mx = m_run;
#pragma unroll
      for (int c = 0; c < 32; ++c) mx = fmaxf(mx, s_sc[w][lane][c]);
      float scale = __expf(m_run - mx);
      float lsum = 0.f;
#pragma unroll
      for (int c = 0; c < 32; ++c) {
        float pv = __expf(s_sc[w][lane][c] - mx);
        lsum += pv;
        s_p[w][lane][c] = (_Float16)pv;
      }
      l_run = l_run * scale + lsum;
      m_run = mx;
      s_aux[w][lane] = scale;
    }
    __syncthreads();
#pragma unroll
    for (int e = 0; e < 8; ++e) {
      float sc = s_aux[w][e + rowoff];
      o0[e] *= sc; o1[e] *= sc; o2[e] *= sc; o3[e] *= sc;
    }
    v16h pa = frag_ld_a(&s_p[w][0][0], 32);
    o0 = wmma16(pa, frag_ld_b(vp + (size_t) 0 * vld + vc0, vld), o0);
    o1 = wmma16(pa, frag_ld_b(vp + (size_t)16 * vld + vc0, vld), o1);
    o2 = wmma16(pa, frag_ld_b(vp + (size_t)32 * vld + vc0, vld), o2);
    o3 = wmma16(pa, frag_ld_b(vp + (size_t)48 * vld + vc0, vld), o3);
    __syncthreads();   // all waves done with tile buffer before refill
#if USE_TDM
    if (w == 0 && kb + 64 < SS) {
      tdm_load_2d(Kb + (size_t)(kb + 64) * DKV, &kt[ib][0][0], 32, 64, DKV);
      tdm_load_2d(Vb + (kb + 64),               &vt[ib][0][0], 64, 32, SS);
    }
#endif
  }
  if (lane < 16) s_aux[w][lane] = 1.0f / l_run;
  __syncthreads();
  int b = bh / NHEAD, hh = bh % NHEAD;
#pragma unroll
  for (int e = 0; e < 8; ++e) {
    int r = e + rowoff;
    float li = s_aux[w][r];
    h16* op = O + ((size_t)(b * SS + q0 + r)) * DMODEL + hh * DKV + colb;
    op[0]  = (h16)(o0[e] * li);
    op[16] = (h16)(o1[e] * li);
    op[32] = (h16)(o2[e] * li);
    op[48] = (h16)(o3[e] * li);
  }
}

// ---------------- stage-2 attention: seq-len 16 (group dim), 1 wave per (s,h) ----------------
__global__ __launch_bounds__(256) void attn2(const h16* __restrict__ Q,
                                             const h16* __restrict__ K,
                                             const h16* __restrict__ Vt,
                                             const float* __restrict__ gmask,
                                             h16* __restrict__ O) {
  __shared__ __align__(16) float    s_sc[8][16][16];
  __shared__ __align__(16) _Float16 s_p[8][16][32];
  __shared__ __align__(16) float    s_aux[8][16];
  int w = threadIdx.x >> 5, lane = threadIdx.x & 31;
  int sh = blockIdx.x * 8 + w;
  const h16*  Qb = Q + (size_t)sh * BB * DKV;
  const h16*  Kb = K + (size_t)sh * BB * DKV;
  const h16*  Vb = Vt + (size_t)sh * DKV * BB;
  const float* Mb = gmask + (size_t)sh * BB * BB;
  int colb = lane & 15, rowoff = (lane >> 4) << 3;
  v16h qa0 = frag_ld_a(Qb, DKV), qa1 = frag_ld_a(Qb + 32, DKV);
  v8f sc = {};
  sc = wmma16(qa0, frag_ld_b(Kb, DKV), sc);
  sc = wmma16(qa1, frag_ld_b(Kb + 32, DKV), sc);
#pragma unroll
  for (int e = 0; e < 8; ++e) {
    int r = e + rowoff;
    s_sc[w][r][colb] = sc[e] + Mb[r * BB + colb];
  }
  __syncthreads();
  if (lane < 16) {
    float mx = -3.0e38f;
#pragma unroll
    for (int c = 0; c < 16; ++c) mx = fmaxf(mx, s_sc[w][lane][c]);
    float lsum = 0.f;
#pragma unroll
    for (int c = 0; c < 16; ++c) {
      float pv = __expf(s_sc[w][lane][c] - mx);
      lsum += pv;
      s_p[w][lane][c] = (_Float16)pv;
    }
#pragma unroll
    for (int c = 16; c < 32; ++c) s_p[w][lane][c] = (_Float16)0.f;
    s_aux[w][lane] = 1.0f / lsum;
  }
  __syncthreads();
  v16h pa = frag_ld_a(&s_p[w][0][0], 32);
  v8f o0 = {}, o1 = {}, o2 = {}, o3 = {};
  o0 = wmma16(pa, frag_ld_b(Vb +  0 * BB, BB), o0);
  o1 = wmma16(pa, frag_ld_b(Vb + 16 * BB, BB), o1);
  o2 = wmma16(pa, frag_ld_b(Vb + 32 * BB, BB), o2);
  o3 = wmma16(pa, frag_ld_b(Vb + 48 * BB, BB), o3);
  int s_idx = sh / NHEAD, hh = sh % NHEAD;
#pragma unroll
  for (int e = 0; e < 8; ++e) {
    int r = e + rowoff;
    float li = s_aux[w][r];
    h16* op = O + ((size_t)(s_idx * BB + r)) * DMODEL + hh * DKV + colb;
    op[0]  = (h16)(o0[e] * li);
    op[16] = (h16)(o1[e] * li);
    op[32] = (h16)(o2[e] * li);
    op[48] = (h16)(o3[e] * li);
  }
}

// ---------------- host launcher ----------------
extern "C" void kernel_launch(void* const* d_in, const int* in_sizes, int n_in,
                              void* d_out, int out_size, void* d_ws, size_t ws_size,
                              hipStream_t stream) {
  (void)in_sizes; (void)n_in; (void)out_size; (void)ws_size;
  const float* hidden   = (const float*)d_in[0];
  const int*   pos      = (const int*)d_in[1];
  const float* amask    = (const float*)d_in[2];
  const float* gmask    = (const float*)d_in[3];
  const float* Wsrc[8]  = {(const float*)d_in[4], (const float*)d_in[5],
                           (const float*)d_in[6], (const float*)d_in[7],
                           (const float*)d_in[9], (const float*)d_in[10],
                           (const float*)d_in[11], (const float*)d_in[12]};
  const float* t_norm   = (const float*)d_in[8];
  const float* g_norm   = (const float*)d_in[13];
  const float* inv_freq = (const float*)d_in[14];
  float* H1 = (float*)d_out;

  char* ws = (char*)d_ws;
  const size_t WELEM = (size_t)DMODEL * DMODEL;
  const size_t XELEM = (size_t)MROWS * DMODEL;
  h16* W16[8];
  for (int i = 0; i < 8; ++i) W16[i] = (h16*)(ws) + i * WELEM;
  size_t off = 8 * WELEM * sizeof(h16);
  h16* X16 = (h16*)(ws + off); off += XELEM * sizeof(h16);
  h16* Q16 = (h16*)(ws + off); off += XELEM * sizeof(h16);
  h16* K16 = (h16*)(ws + off); off += XELEM * sizeof(h16);
  h16* V16 = (h16*)(ws + off); off += XELEM * sizeof(h16) + 256;
  h16* O16 = (h16*)(ws + off);

  for (int i = 0; i < 8; ++i)
    cvt_f16<<<(int)((WELEM + 255) / 256), 256, 0, stream>>>(Wsrc[i], W16[i], (int)WELEM);

  dim3 ggrid(MROWS / 64, DMODEL / 128);

  // ---- stage 1: time attention ----
  rmsnorm_f16<<<MROWS, 256, 0, stream>>>(hidden, t_norm, X16, 0);
  gemm768<<<ggrid, 256, 0, stream>>>(X16, W16[0], 1, SS, nullptr, nullptr, Q16, 0);
  gemm768<<<ggrid, 256, 0, stream>>>(X16, W16[1], 1, SS, nullptr, nullptr, K16, 0);
  gemm768<<<ggrid, 256, 0, stream>>>(X16, W16[2], 2, SS, nullptr, nullptr, V16, 0);
  rope_qk<<<(BB * NHEAD * SS * 32) / 256, 256, 0, stream>>>(Q16, K16, pos, inv_freq);
  attn1<<<dim3(BB * NHEAD, SS / 64), 128, 0, stream>>>(Q16, K16, V16, amask, O16);
  gemm768<<<ggrid, 256, 0, stream>>>(O16, W16[3], 0, SS, hidden, H1, nullptr, 0);

  // ---- stage 2: group attention over batch dim (rows reordered to (s,b)) ----
  rmsnorm_f16<<<MROWS, 256, 0, stream>>>(H1, g_norm, X16, 1);
  gemm768<<<ggrid, 256, 0, stream>>>(X16, W16[4], 1, BB, nullptr, nullptr, Q16, 0);
  gemm768<<<ggrid, 256, 0, stream>>>(X16, W16[5], 1, BB, nullptr, nullptr, K16, 0);
  gemm768<<<ggrid, 256, 0, stream>>>(X16, W16[6], 2, BB, nullptr, nullptr, V16, 0);
  attn2<<<(SS * NHEAD) / 8, 256, 0, stream>>>(Q16, K16, V16, gmask, O16);
  gemm768<<<ggrid, 256, 0, stream>>>(O16, W16[7], 0, BB, H1, (float*)d_out, nullptr, 1);
}
